// SimpleMLPAttention_154618823099
// MI455X (gfx1250) — matrix operand order repaired and checked by reading, not compiled, not run on hardware
//
#include <hip/hip_runtime.h>
#include <hip/hip_bf16.h>
#include <math.h>

// ---------------------------------------------------------------------------
// MI455X (gfx1250) implementation.
// Heavy GEMMs (X@W^T, 5x [8192,1024]x[1024,1024]) run on v_wmma_f32_16x16x32_bf16
// (fp32 -> bf16 in LDS, f32 accumulate) with double-buffered LDS tiles so the
// 8 global b128 loads of tile k+1 overlap the 8 WMMAs of tile k.
// The sequential fast-weight chunk scan stages q/k/v chunks with
// global_load_async_to_lds_b128 (ASYNCcnt DMA path), fp32 VALU math.
// ---------------------------------------------------------------------------

typedef __attribute__((ext_vector_type(16))) __bf16 v16bf;
typedef __attribute__((ext_vector_type(8)))  __bf16 v8bf;
typedef __attribute__((ext_vector_type(8)))  float  v8f;

#define TOKENS 8192      // b*l
#define DIMC   1024
#define HH     16
#define DHD    64
#define DINN   4
#define CHK    16
#define NCHUNK 128
#define NBATCH 4
#define SEQL   2048

// ----------------------------- WMMA GEMM -----------------------------------
// C[M,N] = A[M,K] @ B[N,K]^T   (A,B fp32 in memory; bf16 compute, f32 accum)
#define BM 128
#define BN 128
#define BK 32
#define LPAD 8   // bf16 pad -> row stride 80B (multiple of 16B for v8bf loads)

__global__ __launch_bounds__(256) void gemm_nt_wmma(
    const float* __restrict__ A,
    const float* __restrict__ B0, const float* __restrict__ B1,
    const float* __restrict__ B2, const float* __restrict__ B3,
    float* __restrict__ C0, float* __restrict__ C1,
    float* __restrict__ C2, float* __restrict__ C3,
    int M, int N, int K)
{
  const float* Bp; float* Cp;
  switch (blockIdx.z) {
    case 0:  Bp = B0; Cp = C0; break;
    case 1:  Bp = B1; Cp = C1; break;
    case 2:  Bp = B2; Cp = C2; break;
    default: Bp = B3; Cp = C3; break;
  }

  __shared__ __attribute__((aligned(16))) __bf16 As[2][BM][BK + LPAD];
  __shared__ __attribute__((aligned(16))) __bf16 Bs[2][BN][BK + LPAD];

  const int tid  = threadIdx.x;
  const int lane = tid & 31;
  const int wave = tid >> 5;
  const int wm   = wave & 1;        // 2 waves along M
  const int wn   = wave >> 1;       // 4 waves along N
  const int h16  = lane >> 4;       // lane half (ISA fragment layouts)
  const int r16  = lane & 15;

  const int m0 = blockIdx.y * BM;
  const int n0 = blockIdx.x * BN;

  v8f acc[4][2];
#pragma unroll
  for (int ti = 0; ti < 4; ++ti)
#pragma unroll
    for (int tj = 0; tj < 2; ++tj) {
      v8f z = {0.f, 0.f, 0.f, 0.f, 0.f, 0.f, 0.f, 0.f};
      acc[ti][tj] = z;
    }

  // register staging buffers (tile k+1 loads overlap tile k WMMAs)
  float4 ra[4], rb[4];

  auto issueTile = [&](int k0) {
#pragma unroll
    for (int i = 0; i < 4; ++i) {
      int f   = tid + (i << 8);       // float4 id 0..1023
      int row = f >> 3;               // 8 float4 per 32-wide row
      int col = (f & 7) << 2;
      ra[i] = *(const float4*)(A  + (size_t)(m0 + row) * K + k0 + col);
      rb[i] = *(const float4*)(Bp + (size_t)(n0 + row) * K + k0 + col);
    }
  };
  auto commitTile = [&](int buf) {
#pragma unroll
    for (int i = 0; i < 4; ++i) {
      int f   = tid + (i << 8);
      int row = f >> 3;
      int col = (f & 7) << 2;
      As[buf][row][col + 0] = (__bf16)ra[i].x; As[buf][row][col + 1] = (__bf16)ra[i].y;
      As[buf][row][col + 2] = (__bf16)ra[i].z; As[buf][row][col + 3] = (__bf16)ra[i].w;
      Bs[buf][row][col + 0] = (__bf16)rb[i].x; Bs[buf][row][col + 1] = (__bf16)rb[i].y;
      Bs[buf][row][col + 2] = (__bf16)rb[i].z; Bs[buf][row][col + 3] = (__bf16)rb[i].w;
    }
  };

  issueTile(0);
  commitTile(0);
  __syncthreads();

  int cur = 0;
  for (int k0 = 0; k0 < K; k0 += BK) {
    const bool last = (k0 + BK >= K);
    if (!last) issueTile(k0 + BK);                 // loads in flight over WMMAs
    if (k0 + 2 * BK < K) {                          // hint next-next tile
      int row = tid >> 3, col = (tid & 7) << 2;
      __builtin_prefetch(A  + (size_t)(m0 + row) * K + k0 + 2 * BK + col, 0, 0);
      __builtin_prefetch(Bp + (size_t)(n0 + row) * K + k0 + 2 * BK + col, 0, 0);
    }

    // ---- B fragments: lane(n=r16,half=h16) reads 16 contiguous bf16 at K=16*h16
    v16bf bfrag[2];
#pragma unroll
    for (int tj = 0; tj < 2; ++tj) {
      const __bf16* p = &Bs[cur][wn * 32 + tj * 16 + r16][h16 * 16];
      v8bf lo = *(const v8bf*)(p);
      v8bf hi = *(const v8bf*)(p + 8);
      v16bf b;
#pragma unroll
      for (int e = 0; e < 8; ++e) { b[e] = lo[e]; b[e + 8] = hi[e]; }
      bfrag[tj] = b;
    }

    // ---- A fragments + 8 WMMAs
#pragma unroll
    for (int ti = 0; ti < 4; ++ti) {
      const __bf16* pa = &As[cur][wm * 64 + ti * 16 + r16][0];
      v8bf lo = *(const v8bf*)(pa + h16 * 16);
      v8bf hi = *(const v8bf*)(pa + h16 * 16 + 8);
      v16bf a;
#pragma unroll
      for (int e = 0; e < 8; ++e) { a[e] = lo[e]; a[e + 8] = hi[e]; }
#pragma unroll
      for (int tj = 0; tj < 2; ++tj) {
        acc[ti][tj] = __builtin_amdgcn_wmma_f32_16x16x32_bf16(
            false, a, false, bfrag[tj], (short)0, acc[ti][tj], false, false);
      }
    }

    if (!last) commitTile(cur ^ 1);    // write other buffer (no read conflict)
    __syncthreads();
    cur ^= 1;
  }

  // ---- store C (f32 16x16 D layout: VGPR e -> M = 8*h16 + e, N = r16)
#pragma unroll
  for (int ti = 0; ti < 4; ++ti)
#pragma unroll
    for (int tj = 0; tj < 2; ++tj) {
      int colg = n0 + wn * 32 + tj * 16 + r16;
#pragma unroll
      for (int e = 0; e < 8; ++e) {
        int rowg = m0 + wm * 64 + ti * 16 + h16 * 8 + e;
        Cp[(size_t)rowg * N + colg] = acc[ti][tj][e];
      }
    }
}

// ----------------------------- lr projection -------------------------------
// LRV[token*32 + o] = softplus(X[token] . Wlr[o] + 1e-3),  o = h*2 + which
__global__ __launch_bounds__(256) void lr_kernel(
    const float* __restrict__ X, const float* __restrict__ Wlr,
    float* __restrict__ LRV)
{
  __shared__ __attribute__((aligned(16))) float xs[DIMC];
  const int tok = blockIdx.x;
  for (int i = threadIdx.x; i < DIMC; i += 256)
    xs[i] = X[(size_t)tok * DIMC + i];
  __syncthreads();
  const int wave = threadIdx.x >> 5, lane = threadIdx.x & 31;
  for (int o = wave; o < 2 * HH; o += 8) {
    float acc = 0.f;
    const float* wr = Wlr + (size_t)o * DIMC;
    for (int d = lane; d < DIMC; d += 32) acc += xs[d] * wr[d];
    for (int off = 16; off; off >>= 1) acc += __shfl_down(acc, off);
    if (lane == 0) {
      float x = acc + 1e-3f;
      LRV[(size_t)tok * 32 + o] = (x > 20.f) ? x : log1pf(expf(x));
    }
  }
}

// ----------------------------- short conv ----------------------------------
// y[b,t,c] = x[b,t,c] + sum_j w[c,j] * x[b,t-3+j,c]
__global__ __launch_bounds__(256) void conv_kernel(
    const float* __restrict__ Xq, const float* __restrict__ Xk,
    const float* __restrict__ Xv,
    const float* __restrict__ wq, const float* __restrict__ wk,
    const float* __restrict__ wv,
    float* __restrict__ Yq, float* __restrict__ Yk, float* __restrict__ Yv)
{
  const float* X; const float* w; float* Y;
  switch (blockIdx.z) {
    case 0:  X = Xq; w = wq; Y = Yq; break;
    case 1:  X = Xk; w = wk; Y = Yk; break;
    default: X = Xv; w = wv; Y = Yv; break;
  }
  size_t gid = (size_t)blockIdx.x * 256 + threadIdx.x;  // over TOKENS*DIMC
  int c  = (int)(gid & (DIMC - 1));
  int tl = (int)(gid >> 10);
  int t  = tl & (SEQL - 1);
  int bb = tl >> 11;
  float acc = X[gid];  // identity residual
#pragma unroll
  for (int j = 0; j < 4; ++j) {
    int tt = t - 3 + j;
    if (tt >= 0)
      acc += w[c * 4 + j] * X[((size_t)((bb << 11) + tt) << 10) | c];
  }
  Y[gid] = acc;
}

// ---------------- async memory -> LDS DMA (ASYNCcnt path) -------------------
__device__ __forceinline__ void async_ld_b128(unsigned lds_off,
                                              const float* __restrict__ gbase,
                                              unsigned byte_off)
{
  asm volatile("global_load_async_to_lds_b128 %0, %1, %2"
               :: "v"(lds_off), "v"(byte_off),
                  "s"((unsigned long long)(size_t)gbase)
               : "memory");
}
__device__ __forceinline__ void async_wait0()
{
  asm volatile("s_wait_asynccnt 0x0" ::: "memory");
}

// ----------------------------- chunk scan ----------------------------------
// One block per (batch, head). Sequential over 128 chunks; fp32 in LDS.
__global__ __launch_bounds__(256) void scan_kernel(
    const float* __restrict__ Qc, const float* __restrict__ Kc,
    const float* __restrict__ Vc, const float* __restrict__ LRV,
    const float* __restrict__ WinI, const float* __restrict__ WoutI,
    float* __restrict__ O)
{
  const int bh = blockIdx.x;
  const int bb = bh / HH, h = bh % HH;
  const int tid = threadIdx.x;

  __shared__ __attribute__((aligned(16))) float Win[DINN][DHD];
  __shared__ __attribute__((aligned(16))) float Wout[DINN][DHD];
  __shared__ __attribute__((aligned(16))) float qs[CHK][DHD];
  __shared__ __attribute__((aligned(16))) float ks[CHK][DHD];
  __shared__ __attribute__((aligned(16))) float vs[CHK][DHD];
  __shared__ float sm[CHK][DINN + CHK];  // scores -> probs
  __shared__ float p1[DINN][CHK];
  __shared__ float p2[DINN][CHK];
  __shared__ float lrm[2];

  {
    int n = tid >> 6, d = tid & 63;           // 4*64 = 256 threads
    Win[n][d]  = WinI[((size_t)n * HH + h) * DHD + d];
    Wout[n][d] = WoutI[((size_t)n * HH + h) * DHD + d];
  }

  // per-thread LDS destinations for the async chunk DMA (one b128 each)
  const unsigned lds_q = (unsigned)(size_t)(&qs[0][0]) + (unsigned)tid * 16u;
  const unsigned lds_k = (unsigned)(size_t)(&ks[0][0]) + (unsigned)tid * 16u;
  const unsigned lds_v = (unsigned)(size_t)(&vs[0][0]) + (unsigned)tid * 16u;
  const int st = tid >> 4;            // chunk-local t covered by this thread
  const int sd = (tid & 15) << 2;     // d offset covered by this thread
  __syncthreads();

  const float scale = 0.125f;  // 1/sqrt(64)

  for (int c = 0; c < NCHUNK; ++c) {
    const int t0 = c * CHK;
    // ---- stage q/k/v chunk via async DMA to LDS (16B per thread per tensor)
    {
      unsigned goff = (unsigned)((((bb * SEQL + t0 + st) * DIMC) +
                                  h * DHD + sd) * 4);
      async_ld_b128(lds_q, Qc, goff);
      async_ld_b128(lds_k, Kc, goff);
      async_ld_b128(lds_v, Vc, goff);
      async_wait0();
    }
    __syncthreads();

    // ---- scores s[t][j] and gate logits g1/g2 (+ lr mean)
    for (int i = tid; i < CHK * (DINN + CHK); i += 256) {
      int t = i / (DINN + CHK), j = i % (DINN + CHK);
      const float4* qr = (const float4*)qs[t];
      const float4* kr = (const float4*)((j < DINN) ? Win[j] : ks[j - DINN]);
      float acc = 0.f;
#pragma unroll
      for (int d4 = 0; d4 < DHD / 4; ++d4) {
        float4 a = qr[d4], b = kr[d4];
        acc += a.x * b.x + a.y * b.y + a.z * b.z + a.w * b.w;
      }
      sm[t][j] = acc * scale;
    }
    if (tid < 2 * DINN * CHK) {                 // 128 threads: g1,g2
      int which = tid >> 6;
      int n = (tid >> 4) & 3, cc = tid & 15;
      const float4* wr = (const float4*)Win[n];
      const float4* kr = (const float4*)ks[cc];
      float acc = 0.f;
#pragma unroll
      for (int d4 = 0; d4 < DHD / 4; ++d4) {
        float4 a = wr[d4], b = kr[d4];
        acc += a.x * b.x + a.y * b.y + a.z * b.z + a.w * b.w;
      }
      if (which) p2[n][cc] = acc * (scale * 0.5f);
      else       p1[n][cc] = acc * scale;
    }
    if (tid >= 192 && tid < 194) {              // lr chunk mean (global reads)
      int which = tid - 192;
      float s = 0.f;
      for (int t = 0; t < CHK; ++t)
        s += LRV[((size_t)(bb * SEQL + t0 + t) * HH + h) * 2 + which];
      lrm[which] = s * (1.f / CHK);
    }
    __syncthreads();

    // ---- softmaxes: rows of sm (causal) and rows of p1/p2 (dense)
    if (tid < CHK) {
      int t = tid, lim = t + DINN + 1;
      float mx = -3.0e38f;
      for (int j = 0; j < lim; ++j) mx = fmaxf(mx, sm[t][j]);
      float ssum = 0.f;
      for (int j = 0; j < lim; ++j) { float e = __expf(sm[t][j] - mx); sm[t][j] = e; ssum += e; }
      float inv = 1.f / ssum;
      for (int j = 0; j < lim; ++j) sm[t][j] *= inv;
      for (int j = lim; j < DINN + CHK; ++j) sm[t][j] = 0.f;
    } else if (tid >= 32 && tid < 40) {
      int idx = tid - 32, which = idx >> 2, n = idx & 3;
      float* row = which ? p2[n] : p1[n];
      float mx = -3.0e38f;
      for (int j = 0; j < CHK; ++j) mx = fmaxf(mx, row[j]);
      float ssum = 0.f;
      for (int j = 0; j < CHK; ++j) { float e = __expf(row[j] - mx); row[j] = e; ssum += e; }
      float inv = 1.f / ssum;
      for (int j = 0; j < CHK; ++j) row[j] *= inv;
    }
    __syncthreads();

    // ---- output o = P @ [Wout; v]  and fast-weight grad accumulation
    for (int i = tid; i < CHK * DHD; i += 256) {
      int t = i >> 6, d = i & 63;
      float acc = 0.f;
#pragma unroll
      for (int j = 0; j < DINN; ++j) acc += sm[t][j] * Wout[j][d];
      for (int u = 0; u < CHK; ++u) acc += sm[t][DINN + u] * vs[u][d];
      O[((size_t)(bb * SEQL + t0 + t)) * DIMC + h * DHD + d] = acc;
    }
    float gi = 0.f, go = 0.f;
    {
      int n = tid >> 6, d = tid & 63;
      for (int cc = 0; cc < CHK; ++cc) {
        gi += p2[n][cc] * ks[cc][d];
        go += p1[n][cc] * vs[cc][d];
      }
    }
    __syncthreads();
    {   // W -= lr * Grad, Grad = -softmax@x  =>  W += lr * (p @ x)
      int n = tid >> 6, d = tid & 63;
      Win[n][d]  += lrm[0] * gi;
      Wout[n][d] += lrm[1] * go;
    }
    __syncthreads();
  }
}

// ----------------------------- LayerNorm + gate ----------------------------
// warp per (token, head): LN over DHD, then multiply by gate channel.
__global__ __launch_bounds__(256) void ln_gate_kernel(
    const float* __restrict__ O, const float* __restrict__ G,
    const float* __restrict__ gamma, const float* __restrict__ beta,
    float* __restrict__ Y)
{
  int gw   = (int)((blockIdx.x * 256 + threadIdx.x) >> 5);
  int lane = threadIdx.x & 31;
  int tok = gw >> 4, h = gw & 15;
  const float* row = O + (size_t)tok * DIMC + h * DHD;
  float x0 = row[lane], x1 = row[lane + 32];
  float s = x0 + x1;
  for (int off = 16; off; off >>= 1) s += __shfl_down(s, off);
  s = __shfl(s, 0);
  float mu = s * (1.f / DHD);
  float d0 = x0 - mu, d1 = x1 - mu;
  float v = d0 * d0 + d1 * d1;
  for (int off = 16; off; off >>= 1) v += __shfl_down(v, off);
  v = __shfl(v, 0);
  float inv = rsqrtf(v * (1.f / DHD) + 1e-5f);
  size_t base = (size_t)tok * DIMC;
  int c0 = h * DHD + lane, c1 = c0 + 32;
  Y[base + c0] = (d0 * inv * gamma[lane]      + beta[lane])      * G[base + c0];
  Y[base + c1] = (d1 * inv * gamma[lane + 32] + beta[lane + 32]) * G[base + c1];
}

// ----------------------------- launch --------------------------------------
extern "C" void kernel_launch(void* const* d_in, const int* in_sizes, int n_in,
                              void* d_out, int out_size, void* d_ws, size_t ws_size,
                              hipStream_t stream)
{
  const float* hidden = (const float*)d_in[0];
  const float* Wq     = (const float*)d_in[1];
  const float* Wk     = (const float*)d_in[2];
  const float* Wv     = (const float*)d_in[3];
  const float* Wlr    = (const float*)d_in[4];
  const float* Wo     = (const float*)d_in[5];
  const float* Wg     = (const float*)d_in[6];
  const float* convq  = (const float*)d_in[7];
  const float* convk  = (const float*)d_in[8];
  const float* convv  = (const float*)d_in[9];
  const float* WinI   = (const float*)d_in[10];
  const float* WoutI  = (const float*)d_in[11];
  const float* gamma  = (const float*)d_in[12];
  const float* beta   = (const float*)d_in[13];

  char* ws = (char*)d_ws;
  const size_t MB = (size_t)TOKENS * DIMC * sizeof(float);  // 32 MiB
  float* Qr  = (float*)(ws + 0 * MB);
  float* Kr  = (float*)(ws + 1 * MB);
  float* Vr  = (float*)(ws + 2 * MB);
  float* G   = (float*)(ws + 3 * MB);
  float* Qc  = (float*)(ws + 4 * MB);
  float* Kc  = (float*)(ws + 5 * MB);
  float* Vc  = (float*)(ws + 6 * MB);
  float* LRV = (float*)(ws + 7 * MB);
  float* Ob  = Qr;   // raw Q consumed by conv before scan
  float* Y   = Kr;   // raw K consumed by conv before ln_gate

  // 1) fused projections: Q,K,V,G = X @ {Wq,Wk,Wv,Wg}^T   (WMMA)
  dim3 gp(DIMC / BN, TOKENS / BM, 4);
  gemm_nt_wmma<<<gp, 256, 0, stream>>>(hidden, Wq, Wk, Wv, Wg,
                                       Qr, Kr, Vr, G, TOKENS, DIMC, DIMC);
  // 2) lr = softplus(X @ Wlr^T + base_lr)
  lr_kernel<<<TOKENS, 256, 0, stream>>>(hidden, Wlr, LRV);
  // 3) causal short conv on q,k,v
  dim3 gc((TOKENS * DIMC) / 256, 1, 3);
  conv_kernel<<<gc, 256, 0, stream>>>(Qr, Kr, Vr, convq, convk, convv,
                                      Qc, Kc, Vc);
  // 4) sequential fast-weight chunk scan (one block per (b,h))
  scan_kernel<<<NBATCH * HH, 256, 0, stream>>>(Qc, Kc, Vc, LRV, WinI, WoutI, Ob);
  // 5) LayerNorm + gate
  ln_gate_kernel<<<(TOKENS * HH * 32) / 256, 256, 0, stream>>>(Ob, G, gamma,
                                                               beta, Y);
  // 6) out = Y @ Wo^T   (WMMA)
  dim3 go(DIMC / BN, TOKENS / BM, 1);
  gemm_nt_wmma<<<go, 256, 0, stream>>>(Y, Wo, Wo, Wo, Wo,
                                       (float*)d_out, (float*)d_out,
                                       (float*)d_out, (float*)d_out,
                                       TOKENS, DIMC, DIMC);
}
